// LPManifold_36481452212363
// MI455X (gfx1250) — matrix operand order repaired and checked by reading, not compiled, not run on hardware
//
#include <hip/hip_runtime.h>

#define T_N    8192
#define FEATC  192
#define XCOLS  195          // FEAT + N_WAY + 1
#define OUTC   50
#define KPAD   64           // x1 padded K for WMMA (50 -> 64)
#define KSEL   75
#define TILE_M 16
#define NTILES (T_N / 16)   // 512 column tiles
#define NWAVES 8            // 256-thread block = 8 wave32

typedef _Float16 h16;
typedef __attribute__((ext_vector_type(16))) _Float16 v16h;
typedef __attribute__((ext_vector_type(8)))  _Float16 v8h;
typedef __attribute__((ext_vector_type(8)))  float    v8f;

union H16U { h16 h; unsigned short u; };

// ---------------------------------------------------------------------------
// Kernel 1: x1 = x[:, :192] @ W^T + b  ->  f16 x1h [T][64] (cols 50..63 zero),
//           sq[i] = ||x1_i||^2, x2p = packed x[:,192:195] (stride-4 floats).
// Tiny (157 MFLOP); one 64-thread block per row, row staged through LDS.
// ---------------------------------------------------------------------------
__global__ __launch_bounds__(64)
void lpm_feat_kernel(const float* __restrict__ x, const float* __restrict__ W,
                     const float* __restrict__ b, h16* __restrict__ x1h,
                     float* __restrict__ sq, float* __restrict__ x2p) {
    __shared__ float row[FEATC];
    __shared__ float sqAcc;
    const int r = blockIdx.x;
    const int t = threadIdx.x;
    const float* xr = x + (long)r * XCOLS;
    for (int k = t; k < FEATC; k += 64) row[k] = xr[k];
    if (t == 0) sqAcc = 0.0f;
    __syncthreads();
    if (t < OUTC) {
        const float* w = W + t * FEATC;
        float acc = b[t];
        #pragma unroll 4
        for (int k = 0; k < FEATC; ++k) acc = fmaf(row[k], w[k], acc);
        x1h[(long)r * KPAD + t] = (h16)acc;
        atomicAdd(&sqAcc, acc * acc);
    } else {
        x1h[(long)r * KPAD + t] = (h16)0.0f;   // zero-pad K 50..63
    }
    if (t >= 60) {                             // pack x2 (3 floats, pad to 4)
        x2p[r * 4 + (t - 60)] = (t < 63) ? xr[FEATC + (t - 60)] : 0.0f;
    }
    __syncthreads();
    if (t == 0) sq[r] = sqAcc;
}

// ---------------------------------------------------------------------------
// Kernel 2: fused WMMA distance tile + in-LDS exact radix top-75 + aggregate.
// One block per 16-row stripe; 16x8192 f16 distance keys held in LDS (256KB).
// ---------------------------------------------------------------------------
__global__ __launch_bounds__(256)
void lpm_knn_kernel(const h16* __restrict__ x1h, const float* __restrict__ sq,
                    const float* __restrict__ x2p, float* __restrict__ out) {
    extern __shared__ char smem[];
    unsigned short* keys   = (unsigned short*)smem;                    // [16][8192]
    unsigned*       hist   = (unsigned*)(smem + TILE_M * T_N * 2);     // [8][256]
    float*          accRow = (float*)(hist + NWAVES * 256);            // [16][3]
    unsigned*       tieCnt = (unsigned*)(accRow + TILE_M * 3);         // [16]
    float*          sqI    = (float*)(tieCnt + TILE_M);                // [16]

    const int tid  = threadIdx.x;
    const int wid  = tid >> 5;
    const int lane = tid & 31;
    const int i0   = blockIdx.x * TILE_M;

    if (tid < TILE_M * 3) accRow[tid] = 0.0f;
    if (tid < TILE_M) { sqI[tid] = sq[i0 + tid]; tieCnt[tid] = 0u; }

    // ---- A fragments: loaded once, reused for all 512 column tiles ----
    // A order chosen here (not the ISA table order): lane<16 -> M=lane, K base 0; lane>=16 -> K base 16;
    // v16h elems 0..15 = K cb..cb+15 (contiguous, matches B).
    const h16* arow = x1h + (long)(i0 + (lane & 15)) * KPAD;
    const int  cb0  = (lane < 16) ? 0 : 16;
    v16h a0, a1;
    #pragma unroll
    for (int kk = 0; kk < 2; ++kk) {
        const h16* p = arow + kk * 32 + cb0;
        v8h lo = *(const v8h*)(p);
        v8h hi = *(const v8h*)(p + 8);
        v16h a;
        #pragma unroll
        for (int e = 0; e < 8; ++e) { a[e] = lo[e]; a[8 + e] = hi[e]; }
        if (kk == 0) a0 = a; else a1 = a;
    }
    __syncthreads();

    // ---- Phase 1: distance stripe into LDS as monotone f16 bit-keys ----
    const unsigned short KEY_INF = 0x7C00u;
    for (int ct = wid; ct < NTILES; ct += NWAVES) {
        // B order chosen here (same as A): lane holds column N=lane&15, 16 contiguous K.
        const int  jcol = ct * 16 + (lane & 15);
        const h16* brow = x1h + (long)jcol * KPAD;
        const int  kb   = (lane < 16) ? 0 : 16;
        v16h b0 = *(const v16h*)(brow + kb);
        v16h b1 = *(const v16h*)(brow + 32 + kb);
        float sqj = sq[jcol];
        if (ct + NWAVES < NTILES)  // warm next B tile: global_prefetch_b8
            __builtin_prefetch(x1h + (long)((ct + NWAVES) * 16 + (lane & 15)) * KPAD, 0, 0);

        v8f c = {};
        c = __builtin_amdgcn_wmma_f32_16x16x32_f16(false, a0, false, b0, (short)0, c, false, false);
        c = __builtin_amdgcn_wmma_f32_16x16x32_f16(false, a1, false, b1, (short)0, c, false, false);

        const int mbase = (lane < 16) ? 0 : 8;   // C/D: VGPR v -> M = v (+8)
        #pragma unroll
        for (int v = 0; v < 8; ++v) {
            const int m = mbase + v;
            float d = sqI[m] + sqj - 2.0f * c[v];
            if (d < 0.0f) d = 0.0f;
            H16U ku; ku.h = (h16)d;
            unsigned short key = ku.u;
            if (i0 + m == jcol) key = KEY_INF;   // mask diagonal
            keys[m * T_N + jcol] = key;
        }
    }
    __syncthreads();

    // ---- Phase 2: per-row exact 2-pass radix select (top-75) + aggregate ----
    unsigned* h = hist + wid * 256;
    for (int rr = wid; rr < TILE_M; rr += NWAVES) {
        const unsigned short* krow = keys + rr * T_N;

        // pass 1: high-byte histogram
        for (int bn = lane; bn < 256; bn += 32) h[bn] = 0u;
        for (int j = lane; j < T_N; j += 32)
            atomicAdd(&h[krow[j] >> 8], 1u);
        unsigned cum = 0, b1v = 255, lessHi = 0;
        for (int bn = 0; bn < 256; ++bn) {       // all lanes compute identically
            unsigned nc = cum + h[bn];
            if (cum < KSEL && nc >= KSEL) { b1v = (unsigned)bn; lessHi = cum; }
            cum = nc;
        }

        // pass 2: low-byte histogram within winning high-byte bin
        for (int bn = lane; bn < 256; bn += 32) h[bn] = 0u;
        for (int j = lane; j < T_N; j += 32) {
            unsigned short k = krow[j];
            if ((unsigned)(k >> 8) == b1v) atomicAdd(&h[k & 255u], 1u);
        }
        cum = lessHi; unsigned b2v = 255, less = lessHi;
        for (int bn = 0; bn < 256; ++bn) {
            unsigned nc = cum + h[bn];
            if (cum < KSEL && nc >= KSEL) { b2v = (unsigned)bn; less = cum; }
            cum = nc;
        }
        const unsigned short thr  = (unsigned short)((b1v << 8) | b2v);
        const unsigned       need = KSEL - less; // ties at thr to accept

        // aggregate packed x2 over selected neighbors
        float s0 = 0.0f, s1 = 0.0f, s2 = 0.0f;
        for (int j = lane; j < T_N; j += 32) {
            unsigned short k = krow[j];
            if (k < thr) {
                const float* p = x2p + j * 4;
                s0 += p[0]; s1 += p[1]; s2 += p[2];
            } else if (k == thr) {
                unsigned old = atomicAdd(&tieCnt[rr], 1u);
                if (old < need) {
                    const float* p = x2p + j * 4;
                    s0 += p[0]; s1 += p[1]; s2 += p[2];
                }
            }
        }
        atomicAdd(&accRow[rr * 3 + 0], s0);
        atomicAdd(&accRow[rr * 3 + 1], s1);
        atomicAdd(&accRow[rr * 3 + 2], s2);

        if (lane == 0) {
            const float inv = 1.0f / (float)(KSEL + 1);   // degree is constant 76
            const int   i   = i0 + rr;
            const float* ps = x2p + i * 4;                // self-loop term
            out[i * 3 + 0] = (accRow[rr * 3 + 0] + ps[0]) * inv;
            out[i * 3 + 1] = (accRow[rr * 3 + 1] + ps[1]) * inv;
            out[i * 3 + 2] = (accRow[rr * 3 + 2] + ps[2]) * inv;
        }
    }
}

// ---------------------------------------------------------------------------
extern "C" void kernel_launch(void* const* d_in, const int* in_sizes, int n_in,
                              void* d_out, int out_size, void* d_ws, size_t ws_size,
                              hipStream_t stream) {
    (void)in_sizes; (void)n_in; (void)out_size; (void)ws_size;
    const float* x = (const float*)d_in[0];
    const float* W = (const float*)d_in[1];
    const float* b = (const float*)d_in[2];
    float* out = (float*)d_out;

    h16*   x1h = (h16*)d_ws;                                           // 1 MB
    float* sq  = (float*)((char*)d_ws + (size_t)T_N * KPAD * sizeof(h16));
    float* x2p = sq + T_N;                                             // 128 KB

    lpm_feat_kernel<<<T_N, 64, 0, stream>>>(x, W, b, x1h, sq, x2p);

    const int smem = TILE_M * T_N * 2        // distance keys
                   + NWAVES * 256 * 4       // radix histograms
                   + TILE_M * 3 * 4         // row accumulators
                   + TILE_M * 4             // tie counters
                   + TILE_M * 4;            // sq_i tile
    (void)hipFuncSetAttribute((const void*)lpm_knn_kernel,
                              hipFuncAttributeMaxDynamicSharedMemorySize, smem);
    lpm_knn_kernel<<<T_N / TILE_M, 256, smem, stream>>>(x1h, sq, x2p, out);
}
